// Attention_67525475828089
// MI455X (gfx1250) — matrix operand order repaired and checked by reading, not compiled, not run on hardware
//
#include <hip/hip_runtime.h>

// ---------------------------------------------------------------------------
// Flash attention for:  a = enc . dec^T ; softmax over enc axis ; c = a^T . enc
// i.e. Q = dec, K = V = enc, no scale.  B=8, S=4096, D=256, f32 in/out.
//
// bf16 WMMA (v_wmma_f32_16x16x32_bf16) with f32 accumulation.  enc is streamed
// in 32-row chunks through DOUBLE-BUFFERED LDS in two layouts (row-major for
// the QK^T A-operand, transposed for the PV B-operand) so every WMMA operand
// is a conflict-free, 16B-aligned ds_load_b128.  Global loads for chunk c+1
// are issued into registers while chunk c is computed; one barrier per chunk.
// ---------------------------------------------------------------------------

typedef __attribute__((ext_vector_type(16))) __bf16       v16bf;
typedef __attribute__((ext_vector_type(8)))  float        v8f;
typedef __attribute__((ext_vector_type(8)))  unsigned int v8u;
typedef __attribute__((ext_vector_type(4)))  unsigned int v4u;

#define S_ENC      4096
#define S_DEC      4096
#define DIM        256
#define CHUNK      32     // enc rows staged per iteration
#define KV_STRIDE  264    // ushort stride of [i][d] LDS copy (pad: conflict-free)
#define KVT_STRIDE 40     // ushort stride of [d][i] LDS copy (pad: 16B-aligned rows)

__device__ __forceinline__ unsigned int pack_bf16(float x, float y) {
  unsigned int a = __builtin_bit_cast(unsigned int, x);
  unsigned int b = __builtin_bit_cast(unsigned int, y);
  a = (a + 0x7FFFu + ((a >> 16) & 1u)) >> 16;          // RNE to bf16, low half
  b = (b + 0x7FFFu + ((b >> 16) & 1u)) & 0xFFFF0000u;  // RNE to bf16, high half
  return b | a;
}

__global__ __launch_bounds__(256)
void attn_fa_kernel(const float* __restrict__ enc,
                    const float* __restrict__ dec,
                    float* __restrict__ out) {
  __shared__ unsigned short ldsKV [2 * CHUNK * KV_STRIDE];  // enc [i][d], bf16, x2 buf
  __shared__ unsigned short ldsKVT[2 * DIM   * KVT_STRIDE]; // enc [d][i], bf16, x2 buf

  const int tid  = threadIdx.x;
  const int lane = tid & 31;
  const int wave = tid >> 5;
  const int lm   = lane & 15;   // j (GEMM1 B / GEMM2 A row), or d-in-tile
  const int hi   = lane >> 4;   // lane half selects K sub-range

  const int blocksPerBatch = S_DEC / 128;
  const int b  = blockIdx.x / blocksPerBatch;
  const int j0 = (blockIdx.x % blocksPerBatch) * 128 + wave * 16;

  // ---- Q tile (16 dec rows) -> bf16 B-fragments, kept in VGPRs ----
  // B layout (32x16, 16-bit): lane n=lm, K = 8*hi + {0..7} U 16 + 8*hi + {0..7},
  // VGPRs 0..3 pack K = 8*hi + 0..7, VGPRs 4..7 pack K = 16 + 8*hi + 0..7.
  const float* qrow = dec + ((size_t)b * S_DEC + (size_t)(j0 + lm)) * DIM;
  v16bf qf[8];
#pragma unroll
  for (int dc = 0; dc < 8; ++dc) {
    const float4* qp = (const float4*)(qrow + dc * 32 + hi * 8);
    float4 f0 = qp[0], f1 = qp[1], f2 = qp[4], f3 = qp[5];
    v8u q;
    q[0] = pack_bf16(f0.x, f0.y); q[1] = pack_bf16(f0.z, f0.w);
    q[2] = pack_bf16(f1.x, f1.y); q[3] = pack_bf16(f1.z, f1.w);
    q[4] = pack_bf16(f2.x, f2.y); q[5] = pack_bf16(f2.z, f2.w);
    q[6] = pack_bf16(f3.x, f3.y); q[7] = pack_bf16(f3.z, f3.w);
    qf[dc] = __builtin_bit_cast(v16bf, q);
  }

  // O accumulators: 16 d-tiles of 16x16 f32 (C layout: row j = v + 8*hi)
  v8f o[16];
#pragma unroll
  for (int t = 0; t < 16; ++t)
#pragma unroll
    for (int v = 0; v < 8; ++v) o[t][v] = 0.f;

  float m = -3.0e38f;  // running max per column j = lm (replicated in lane halves)
  float l = 0.f;       // running denominator

  const float* ebase = enc + (size_t)b * S_ENC * DIM;
  const int nch = S_ENC / CHUNK;

  // ---- prefetch chunk 0 into registers ----
  float4 pref[8];
#pragma unroll
  for (int r = 0; r < 8; ++r)
    pref[r] = *(const float4*)(ebase + r * 1024 + tid * 4);

  for (int c = 0; c < nch; ++c) {
    unsigned short* KVb  = ldsKV  + (c & 1) * (CHUNK * KV_STRIDE);
    unsigned short* KVTb = ldsKVT + (c & 1) * (DIM * KVT_STRIDE);

    // ---- convert + store staged chunk (32 x 256) to both LDS layouts ----
#pragma unroll
    for (int r = 0; r < 8; ++r) {
      int idx = r * 1024 + tid * 4;          // 8192 elems / 256 threads
      int row = idx >> 8, col = idx & 255;
      unsigned int p0 = pack_bf16(pref[r].x, pref[r].y);
      unsigned int p1 = pack_bf16(pref[r].z, pref[r].w);
      *(uint2*)&KVb[row * KV_STRIDE + col] = make_uint2(p0, p1);
      KVTb[(col + 0) * KVT_STRIDE + row] = (unsigned short)(p0 & 0xFFFFu);
      KVTb[(col + 1) * KVT_STRIDE + row] = (unsigned short)(p0 >> 16);
      KVTb[(col + 2) * KVT_STRIDE + row] = (unsigned short)(p1 & 0xFFFFu);
      KVTb[(col + 3) * KVT_STRIDE + row] = (unsigned short)(p1 >> 16);
    }

    // ---- issue global loads for chunk c+1 (in flight across the barrier) ----
    if (c + 1 < nch) {
      const float* nb = ebase + (size_t)(c + 1) * CHUNK * DIM;
#pragma unroll
      for (int r = 0; r < 8; ++r)
        pref[r] = *(const float4*)(nb + r * 1024 + tid * 4);
    }

    __syncthreads();  // buffer (c&1) fully staged; buffer ((c+1)&1) free

    // ---- GEMM1: S[i x j], two 16x16 tiles (i local 0..15, 16..31) ----
    v8f s0, s1;
#pragma unroll
    for (int v = 0; v < 8; ++v) { s0[v] = 0.f; s1[v] = 0.f; }
#pragma unroll
    for (int dc = 0; dc < 8; ++dc) {
      // A layout (16x32): lane row = lm, K = dc*32 + 8*hi + {0..7} U {16..23}
      const unsigned short* a0 = &KVb[(lm)      * KV_STRIDE + dc * 32 + hi * 8];
      const unsigned short* a1 = &KVb[(16 + lm) * KV_STRIDE + dc * 32 + hi * 8];
      v4u x0 = *(const v4u*)a0, y0 = *(const v4u*)(a0 + 16);
      v4u x1 = *(const v4u*)a1, y1 = *(const v4u*)(a1 + 16);
      v8u au0, au1;
#pragma unroll
      for (int v = 0; v < 4; ++v) {
        au0[v] = x0[v]; au0[v + 4] = y0[v];
        au1[v] = x1[v]; au1[v + 4] = y1[v];
      }
      s0 = __builtin_amdgcn_wmma_f32_16x16x32_bf16(
          false, __builtin_bit_cast(v16bf, au0), false, qf[dc], (short)0, s0, false, false);
      s1 = __builtin_amdgcn_wmma_f32_16x16x32_bf16(
          false, __builtin_bit_cast(v16bf, au1), false, qf[dc], (short)0, s1, false, false);
    }

    // ---- online softmax over i (column j = lm per lane) ----
    float cm = s0[0];
#pragma unroll
    for (int v = 1; v < 8; ++v) cm = fmaxf(cm, s0[v]);
#pragma unroll
    for (int v = 0; v < 8; ++v) cm = fmaxf(cm, s1[v]);
    cm = fmaxf(cm, __shfl_xor(cm, 16, 32));       // combine lane halves (i rows)
    float mn    = fmaxf(m, cm);
    float alpha = __expf(m - mn);                  // == 1.0 exactly when mn == m
    float rsum  = 0.f;
    v8f p0, p1;
#pragma unroll
    for (int v = 0; v < 8; ++v) {
      p0[v] = __expf(s0[v] - mn);
      p1[v] = __expf(s1[v] - mn);
      rsum += p0[v] + p1[v];
    }
    rsum += __shfl_xor(rsum, 16, 32);
    l = l * alpha + rsum;
    m = mn;

    // ---- pack P^T into bf16 A-fragment: pure per-lane pack (layout match) ----
    v8u pu;
#pragma unroll
    for (int v = 0; v < 4; ++v) {
      pu[v]     = pack_bf16(p0[2 * v], p0[2 * v + 1]);
      pu[v + 4] = pack_bf16(p1[2 * v], p1[2 * v + 1]);
    }
    v16bf pf = __builtin_bit_cast(v16bf, pu);

    // ---- rescale O rows by alpha_j, skipped wave-uniformly when max unchanged
    //      (uniform branch keeps EXEC all-ones for the WMMAs below) ----
    if (__any(alpha < 1.0f)) {
      float ar[8];
#pragma unroll
      for (int v = 0; v < 8; ++v) ar[v] = __shfl(alpha, v + 8 * hi, 32);
#pragma unroll
      for (int t = 0; t < 16; ++t)
#pragma unroll
        for (int v = 0; v < 8; ++v) o[t][v] *= ar[v];
    }

    // ---- GEMM2: O[j x d] += P^T x encChunk, B from transposed LDS copy ----
#pragma unroll
    for (int t = 0; t < 16; ++t) {
      // B layout: lane n = d = t*16+lm, K = i local 8*hi + {0..7} U {16..23}
      const unsigned short* bp = &KVTb[(t * 16 + lm) * KVT_STRIDE + hi * 8];
      v4u x = *(const v4u*)bp, y = *(const v4u*)(bp + 16);
      v8u bu;
#pragma unroll
      for (int v = 0; v < 4; ++v) { bu[v] = x[v]; bu[v + 4] = y[v]; }
      o[t] = __builtin_amdgcn_wmma_f32_16x16x32_bf16(
          false, pf, false, __builtin_bit_cast(v16bf, bu), (short)0, o[t], false, false);
    }
  }

  // ---- finalize: divide by softmax denominator and store f32 ----
  float linv = 1.f / l;
  float lr[8];
#pragma unroll
  for (int v = 0; v < 8; ++v) lr[v] = __shfl(linv, v + 8 * hi, 32);
  float* obase = out + ((size_t)b * S_DEC + (size_t)j0) * DIM;
#pragma unroll
  for (int t = 0; t < 16; ++t)
#pragma unroll
    for (int v = 0; v < 8; ++v)
      obase[(size_t)(v + 8 * hi) * DIM + t * 16 + lm] = o[t][v] * lr[v];
}

extern "C" void kernel_launch(void* const* d_in, const int* in_sizes, int n_in,
                              void* d_out, int out_size, void* d_ws, size_t ws_size,
                              hipStream_t stream) {
  (void)in_sizes; (void)n_in; (void)d_ws; (void)ws_size; (void)out_size;
  const float* enc = (const float*)d_in[0];
  const float* dec = (const float*)d_in[1];
  float* out = (float*)d_out;
  dim3 grid(8 * (S_DEC / 128));  // B * j-blocks = 256 workgroups
  attn_fa_kernel<<<grid, dim3(256), 0, stream>>>(enc, dec, out);
}
